// KMeans_83124797047435
// MI455X (gfx1250) — matrix operand order repaired and checked by reading, not compiled, not run on hardware
//
#include <hip/hip_runtime.h>

#define N_PTS 131072
#define DIM   64
#define KCB   1024
#define EPSF  1e-5f

typedef __attribute__((ext_vector_type(16))) __bf16 v16bf;
typedef __attribute__((ext_vector_type(8)))  float  v8f;

// ---------------------------------------------------------------------------
// Kernel 1: split embedding into bf16 hi/lo pair and compute ||e_k||^2.
// One wave32 per codeword; each lane handles d=lane and d=lane+32.
// ---------------------------------------------------------------------------
__global__ void __launch_bounds__(256)
vq_prep(const float* __restrict__ emb,
        float* __restrict__ e_norm,
        __bf16* __restrict__ eh,
        __bf16* __restrict__ el) {
    int wave = (blockIdx.x * blockDim.x + threadIdx.x) >> 5;  // codeword id
    int lane = threadIdx.x & 31;
    if (wave >= KCB) return;
    const float* rp = emb + wave * DIM;
    float f0 = rp[lane], f1 = rp[lane + 32];
    __bf16 h0 = (__bf16)f0, h1 = (__bf16)f1;
    eh[wave * DIM + lane]      = h0;
    eh[wave * DIM + lane + 32] = h1;
    el[wave * DIM + lane]      = (__bf16)(f0 - (float)h0);
    el[wave * DIM + lane + 32] = (__bf16)(f1 - (float)h1);
    float s = f0 * f0 + f1 * f1;
    #pragma unroll
    for (int off = 16; off >= 1; off >>= 1) s += __shfl_xor(s, off, 32);
    if (lane == 0) e_norm[wave] = s;
}

// ---------------------------------------------------------------------------
// Kernel 2: assignment via WMMA (bf16 hi/lo split GEMM), argmin, and
// atomic segment-sum scatter. 8 waves/block, 16 x-rows per wave.
// ---------------------------------------------------------------------------
__global__ void __launch_bounds__(256)
vq_assign(const float* __restrict__ x,
          const float* __restrict__ e_norm,
          const __bf16* __restrict__ eh,
          const __bf16* __restrict__ el,
          float* __restrict__ count,
          float* __restrict__ weight,
          float* __restrict__ dist_sum) {
    __shared__ float xn[8][16];   // per-wave row norms
    __shared__ int   sidx[8][16]; // per-wave row argmin indices

    const int w    = threadIdx.x >> 5;
    const int lane = threadIdx.x & 31;
    const int half = lane >> 4;   // 0: lanes 0-15, 1: lanes 16-31
    const int l16  = lane & 15;

    const int rowBase = blockIdx.x * 128 + w * 16;   // first of 16 rows
    const float* xr = x + (size_t)(rowBase + l16) * DIM;   // this lane's row

    // ---- Build A fragments (x tile), 2 K-blocks of 32, hi/lo split.
    // A k order used here (matches B): element j -> K = kb*32 + half*16 + j
    v16bf Ah[2], Al[2];
    float normPart = 0.f;
    #pragma unroll
    for (int kb = 0; kb < 2; ++kb) {
        const int b0 = kb * 32 + half * 16;
        const int b1 = kb * 32 + half * 16 + 8;
        #pragma unroll
        for (int j = 0; j < 8; ++j) {
            float f = xr[b0 + j];
            __bf16 h = (__bf16)f;
            Ah[kb][j] = h;
            Al[kb][j] = (__bf16)(f - (float)h);
            normPart += f * f;
            float g = xr[b1 + j];
            __bf16 hg = (__bf16)g;
            Ah[kb][8 + j] = hg;
            Al[kb][8 + j] = (__bf16)(g - (float)hg);
            normPart += g * g;
        }
    }
    // lanes l16 and l16+16 hold complementary K-halves of the same row
    normPart += __shfl_xor(normPart, 16, 32);
    if (lane < 16) xn[w][lane] = normPart;

    float minv[8];
    int   mini[8];
    #pragma unroll
    for (int i = 0; i < 8; ++i) { minv[i] = 3.4e38f; mini[i] = 0; }

    // ---- Sweep all K=1024 codewords in 64 tiles of 16.
    for (int t = 0; t < 64; ++t) {
        const int c = t * 16 + l16;  // codeword column owned by this lane
        const __bf16* ehp = eh + (size_t)c * DIM;
        const __bf16* elp = el + (size_t)c * DIM;
        // B k order used here (KxN, matches A): lane = column n; element j -> K = kb*32 + half*16 + j
        v16bf Bh0 = *(const v16bf*)(ehp + half * 16);
        v16bf Bl0 = *(const v16bf*)(elp + half * 16);
        v16bf Bh1 = *(const v16bf*)(ehp + 32 + half * 16);
        v16bf Bl1 = *(const v16bf*)(elp + 32 + half * 16);
        const float en = e_norm[c];

        v8f acc = {};
        // dot ~= xh*eh + xh*el + xl*eh  (xl*el term ~2^-32, dropped)
        acc = __builtin_amdgcn_wmma_f32_16x16x32_bf16(false, Ah[0], false, Bh0, (short)0, acc, false, false);
        acc = __builtin_amdgcn_wmma_f32_16x16x32_bf16(false, Ah[0], false, Bl0, (short)0, acc, false, false);
        acc = __builtin_amdgcn_wmma_f32_16x16x32_bf16(false, Al[0], false, Bh0, (short)0, acc, false, false);
        acc = __builtin_amdgcn_wmma_f32_16x16x32_bf16(false, Ah[1], false, Bh1, (short)0, acc, false, false);
        acc = __builtin_amdgcn_wmma_f32_16x16x32_bf16(false, Ah[1], false, Bl1, (short)0, acc, false, false);
        acc = __builtin_amdgcn_wmma_f32_16x16x32_bf16(false, Al[1], false, Bh1, (short)0, acc, false, false);

        // C/D layout: VGPR i, lanes 0-15 -> row i; lanes 16-31 -> row i+8
        #pragma unroll
        for (int i = 0; i < 8; ++i) {
            float d = en - 2.0f * acc[i];   // ||e||^2 - 2 x.e  (||x||^2 added later)
            if (d < minv[i]) { minv[i] = d; mini[i] = c; }
        }
    }

    // ---- Cross-lane argmin within each 16-lane half (columns of the tile).
    #pragma unroll
    for (int i = 0; i < 8; ++i) {
        float v = minv[i]; int id = mini[i];
        #pragma unroll
        for (int off = 8; off >= 1; off >>= 1) {
            float ov = __shfl_xor(v, off, 32);
            int   oi = __shfl_xor(id, off, 32);
            if (ov < v || (ov == v && oi < id)) { v = ov; id = oi; }
        }
        minv[i] = v; mini[i] = id;
    }

    // lanes 0 / 16 publish rows 0..7 / 8..15; accumulate distortion.
    if (l16 == 0) {
        float dsum = 0.f;
        #pragma unroll
        for (int i = 0; i < 8; ++i) {
            int row = i + half * 8;
            sidx[w][row] = mini[i];
            dsum += minv[i] + xn[w][row];   // full squared distance
        }
        atomicAdd(dist_sum, dsum);
    }
    __syncthreads();

    // ---- Scatter segment sums: 16 rows x 64 cols; 2 cols per lane.
    for (int r = 0; r < 16; ++r) {
        const int idx = sidx[w][r];
        const float* rp = x + (size_t)(rowBase + r) * DIM;
        atomicAdd(&weight[(size_t)idx * DIM + lane],      rp[lane]);
        atomicAdd(&weight[(size_t)idx * DIM + lane + 32], rp[lane + 32]);
        if (lane == 0) atomicAdd(&count[idx], 1.0f);
    }
}

// ---------------------------------------------------------------------------
// Kernel 3: finalize.  n = sum(count) == N exactly (every point assigned).
// out = [distortion(1) | new_embedding(K*64) | ema_count(K)]
// ---------------------------------------------------------------------------
__global__ void __launch_bounds__(256)
vq_final(const float* __restrict__ count,
         const float* __restrict__ weight,
         const float* __restrict__ dist_sum,
         float* __restrict__ out) {
    int t = blockIdx.x * blockDim.x + threadIdx.x;
    if (t >= KCB * DIM) return;
    int k = t >> 6, d = t & 63;
    const float nf = (float)N_PTS;
    float c = (count[k] + EPSF) / (nf + (float)KCB * EPSF) * nf;
    out[1 + t] = weight[t] / c;
    if (d == 0) out[1 + KCB * DIM + k] = c;
    if (t == 0) out[0] = dist_sum[0] / (nf * (float)DIM);
}

// ---------------------------------------------------------------------------
extern "C" void kernel_launch(void* const* d_in, const int* in_sizes, int n_in,
                              void* d_out, int out_size, void* d_ws, size_t ws_size,
                              hipStream_t stream) {
    (void)in_sizes; (void)n_in; (void)out_size;
    const float* x   = (const float*)d_in[1];   // [N, 64]
    const float* emb = (const float*)d_in[2];   // [K, 64]
    float* out = (float*)d_out;

    // Workspace layout (floats unless noted):
    //   count[K] | weight[K*64] | dist_sum[1] | pad(3) | e_norm[K] | eh[K*64 bf16] | el[K*64 bf16]
    float*  ws_count  = (float*)d_ws;
    float*  ws_weight = ws_count + KCB;
    float*  ws_dsum   = ws_weight + KCB * DIM;
    float*  ws_enorm  = ws_dsum + 4;            // 16B-aligned
    __bf16* ws_eh     = (__bf16*)(ws_enorm + KCB);
    __bf16* ws_el     = ws_eh + KCB * DIM;

    // Zero the accumulators (count + weight + dist_sum region).
    hipMemsetAsync(d_ws, 0, (size_t)(KCB + KCB * DIM + 1) * sizeof(float), stream);

    // 1) embedding split + norms: K waves -> K/8 = 128 blocks of 256.
    vq_prep<<<KCB / 8, 256, 0, stream>>>(emb, ws_enorm, ws_eh, ws_el);

    // 2) assignment: 131072 rows / (8 waves * 16 rows) = 1024 blocks.
    vq_assign<<<N_PTS / 128, 256, 0, stream>>>(x, ws_enorm, ws_eh, ws_el,
                                               ws_count, ws_weight, ws_dsum);

    // 3) finalize: K*64 threads.
    vq_final<<<(KCB * DIM) / 256, 256, 0, stream>>>(ws_count, ws_weight, ws_dsum, out);
}
